// ShareFeatureSplitComplementBlock1_34342558499508
// MI455X (gfx1250) — matrix operand order repaired and checked by reading, not compiled, not run on hardware
//
#include <hip/hip_runtime.h>
#include <hip/hip_bf16.h>

// Problem constants from the reference.
#define DD 512
#define SS 16
#define TT 512
#define NN 8192

typedef __attribute__((ext_vector_type(16))) __bf16 v16bf;
typedef __attribute__((ext_vector_type(8)))  __bf16 v8bf;
typedef __attribute__((ext_vector_type(8)))  float  v8f;

// ---------------------------------------------------------------------------
// WMMA fragment helpers (v_wmma_f32_16x16x32_bf16, wave32).
// A layout (16x32, MxK): lane half 0 -> K {0..7,16..23}, half 1 -> {8..15,24..31}
// B layout (32x16, KxN): lane half 0 -> K {0..7,16..23}, half 1 -> {8..15,24..31}
// Both operands are row-major over k (NT GEMM), so all loads are contiguous.
// ---------------------------------------------------------------------------
__device__ __forceinline__ v16bf load_frag_a(const __bf16* base, int ld, int kb, int lane) {
  const int half = lane >> 4;
  const int row  = lane & 15;
  const __bf16* p = base + (size_t)row * ld + kb + half * 8;
  v8bf lo = *(const v8bf*)p;
  v8bf hi = *(const v8bf*)(p + 16);
  v16bf r;
#pragma unroll
  for (int i = 0; i < 8; ++i) { r[i] = lo[i]; r[i + 8] = hi[i]; }
  return r;
}

__device__ __forceinline__ v16bf load_frag_b(const __bf16* base, int ld, int kb, int lane) {
  const int half = lane >> 4;
  const int col  = lane & 15;
  const __bf16* p = base + (size_t)col * ld + kb + half * 8;
  v8bf lo = *(const v8bf*)p;
  v8bf hi = *(const v8bf*)(p + 16);
  v16bf r;
#pragma unroll
  for (int i = 0; i < 8; ++i) { r[i] = lo[i]; r[i + 8] = hi[i]; }
  return r;
}

__device__ __forceinline__ v8f wmma_bf16(v16bf a, v16bf b, v8f c) {
  return __builtin_amdgcn_wmma_f32_16x16x32_bf16(false, a, false, b, (short)0, c, false, false);
}

#define ZERO8 {0.f, 0.f, 0.f, 0.f, 0.f, 0.f, 0.f, 0.f}

// ---------------------------------------------------------------------------
// NT GEMM kernels: one wave per 16(M) x 64(N) tile (4 accumulators sharing the
// A fragment), 8 waves / 256-thread block. lda == ldb == K in every use.
// Grids are sized so (M/16)*(N/64) is a multiple of 8.
// ---------------------------------------------------------------------------
__global__ void gemm_nt_bf16(const __bf16* __restrict__ A, const __bf16* __restrict__ B,
                             __bf16* __restrict__ C, int M, int Nn, int K, float scale,
                             size_t strideA, size_t strideC) {
  const int wave = threadIdx.x >> 5;
  const int lane = threadIdx.x & 31;
  const int tilesN64 = Nn >> 6;
  const int tile = blockIdx.x * 8 + wave;
  const int tm = tile / tilesN64;
  const int tn = (tile - tm * tilesN64) * 4;     // in 16-col units
  A += (size_t)blockIdx.y * strideA;
  C += (size_t)blockIdx.y * strideC;
  const __bf16* Ar = A + (size_t)(tm << 4) * K;
  const __bf16* Br = B + (size_t)(tn << 4) * K;
  v8f acc[4] = {ZERO8, ZERO8, ZERO8, ZERO8};
  for (int kb = 0; kb < K; kb += 32) {
    __builtin_prefetch(Ar + (size_t)(lane & 15) * K + kb + 32, 0, 1);
    v16bf a = load_frag_a(Ar, K, kb, lane);
#pragma unroll
    for (int j = 0; j < 4; ++j) {
      v16bf b = load_frag_b(Br + (size_t)(j << 4) * K, K, kb, lane);
      acc[j] = wmma_bf16(a, b, acc[j]);
    }
  }
  const int half = lane >> 4, col = lane & 15;
#pragma unroll
  for (int j = 0; j < 4; ++j)
#pragma unroll
    for (int i = 0; i < 8; ++i)
      C[(size_t)((tm << 4) + half * 8 + i) * Nn + ((tn + j) << 4) + col] =
          (__bf16)(acc[j][i] * scale);
}

// Same GEMM but stores C transposed (used to produce V^T[D,N]).
__global__ void gemm_nt_bf16_tc(const __bf16* __restrict__ A, const __bf16* __restrict__ B,
                                __bf16* __restrict__ C, int M, int Nn, int K) {
  const int wave = threadIdx.x >> 5;
  const int lane = threadIdx.x & 31;
  const int tilesN64 = Nn >> 6;
  const int tile = blockIdx.x * 8 + wave;
  const int tm = tile / tilesN64;
  const int tn = (tile - tm * tilesN64) * 4;
  const __bf16* Ar = A + (size_t)(tm << 4) * K;
  const __bf16* Br = B + (size_t)(tn << 4) * K;
  v8f acc[4] = {ZERO8, ZERO8, ZERO8, ZERO8};
  for (int kb = 0; kb < K; kb += 32) {
    __builtin_prefetch(Ar + (size_t)(lane & 15) * K + kb + 32, 0, 1);
    v16bf a = load_frag_a(Ar, K, kb, lane);
#pragma unroll
    for (int j = 0; j < 4; ++j) {
      v16bf b = load_frag_b(Br + (size_t)(j << 4) * K, K, kb, lane);
      acc[j] = wmma_bf16(a, b, acc[j]);
    }
  }
  const int half = lane >> 4, col = lane & 15;
#pragma unroll
  for (int j = 0; j < 4; ++j)
#pragma unroll
    for (int i = 0; i < 8; ++i)
      C[(size_t)(((tn + j) << 4) + col) * M + (tm << 4) + half * 8 + i] = (__bf16)acc[j][i];
}

// NT GEMM + bias, f32 output (pre-LayerNorm activations).
__global__ void gemm_nt_f32_bias(const __bf16* __restrict__ A, const __bf16* __restrict__ B,
                                 const float* __restrict__ bias, float* __restrict__ C,
                                 int M, int Nn, int K) {
  const int wave = threadIdx.x >> 5;
  const int lane = threadIdx.x & 31;
  const int tilesN64 = Nn >> 6;
  const int tile = blockIdx.x * 8 + wave;
  const int tm = tile / tilesN64;
  const int tn = (tile - tm * tilesN64) * 4;
  const __bf16* Ar = A + (size_t)(tm << 4) * K;
  const __bf16* Br = B + (size_t)(tn << 4) * K;
  v8f acc[4] = {ZERO8, ZERO8, ZERO8, ZERO8};
  for (int kb = 0; kb < K; kb += 32) {
    __builtin_prefetch(Ar + (size_t)(lane & 15) * K + kb + 32, 0, 1);
    v16bf a = load_frag_a(Ar, K, kb, lane);
#pragma unroll
    for (int j = 0; j < 4; ++j) {
      v16bf b = load_frag_b(Br + (size_t)(j << 4) * K, K, kb, lane);
      acc[j] = wmma_bf16(a, b, acc[j]);
    }
  }
  const int half = lane >> 4, col = lane & 15;
#pragma unroll
  for (int j = 0; j < 4; ++j) {
    const float bv = bias[((tn + j) << 4) + col];
#pragma unroll
    for (int i = 0; i < 8; ++i)
      C[(size_t)((tm << 4) + half * 8 + i) * Nn + ((tn + j) << 4) + col] = acc[j][i] + bv;
  }
}

// NT GEMM + bias, dual f32 + bf16 output (final kqv@Wm+bm -> out0 + Mb).
__global__ void gemm_nt_f32bf16_bias(const __bf16* __restrict__ A, const __bf16* __restrict__ B,
                                     const float* __restrict__ bias, float* __restrict__ Cf,
                                     __bf16* __restrict__ Cb, int M, int Nn, int K) {
  const int wave = threadIdx.x >> 5;
  const int lane = threadIdx.x & 31;
  const int tilesN64 = Nn >> 6;
  const int tile = blockIdx.x * 8 + wave;
  const int tm = tile / tilesN64;
  const int tn = (tile - tm * tilesN64) * 4;
  const __bf16* Ar = A + (size_t)(tm << 4) * K;
  const __bf16* Br = B + (size_t)(tn << 4) * K;
  v8f acc[4] = {ZERO8, ZERO8, ZERO8, ZERO8};
  for (int kb = 0; kb < K; kb += 32) {
    __builtin_prefetch(Ar + (size_t)(lane & 15) * K + kb + 32, 0, 1);
    v16bf a = load_frag_a(Ar, K, kb, lane);
#pragma unroll
    for (int j = 0; j < 4; ++j) {
      v16bf b = load_frag_b(Br + (size_t)(j << 4) * K, K, kb, lane);
      acc[j] = wmma_bf16(a, b, acc[j]);
    }
  }
  const int half = lane >> 4, col = lane & 15;
#pragma unroll
  for (int j = 0; j < 4; ++j) {
    const float bv = bias[((tn + j) << 4) + col];
#pragma unroll
    for (int i = 0; i < 8; ++i) {
      const size_t idx = (size_t)((tm << 4) + half * 8 + i) * Nn + ((tn + j) << 4) + col;
      const float v = acc[j][i] + bv;
      Cf[idx] = v;
      Cb[idx] = (__bf16)v;
    }
  }
}

// ---------------------------------------------------------------------------
// Attention logits + full-row softmax. Softmax axis is t (T=512), so a
// [16 rows x 512 cols] tile in LDS holds complete softmax rows: no online
// rescaling. 8 waves compute the tile with WMMA (each wave: one A fragment
// reused across 4 column tiles); then 16 threads/row reduce.
// Output written as attn^T in bf16: A2[s][t, n].
// ---------------------------------------------------------------------------
__global__ void attn_softmax_kernel(const __bf16* __restrict__ Kb, const __bf16* __restrict__ Qb,
                                    __bf16* __restrict__ A2) {
  const int s    = blockIdx.y;
  const int nblk = blockIdx.x;             // 16-row block of N
  const int tid  = threadIdx.x;            // 256 threads = 8 waves
  const int wave = tid >> 5;
  const int lane = tid & 31;
  __shared__ float Ltile[16][TT + 4];
  __shared__ float red[16][16];

  const __bf16* Qs   = Qb + (size_t)s * TT * DD + (size_t)(wave * 64) * DD;
  const __bf16* Arow = Kb + (size_t)(nblk << 4) * DD;
  const int half = lane >> 4, col = lane & 15;

  v8f acc[4] = {ZERO8, ZERO8, ZERO8, ZERO8};
  for (int kb = 0; kb < DD; kb += 32) {
    __builtin_prefetch(Arow + (size_t)(lane & 15) * DD + kb + 32, 0, 1);
    v16bf a = load_frag_a(Arow, DD, kb, lane);
#pragma unroll
    for (int c = 0; c < 4; ++c) {
      v16bf b = load_frag_b(Qs + (size_t)(c << 4) * DD, DD, kb, lane);
      acc[c] = wmma_bf16(a, b, acc[c]);
    }
  }
#pragma unroll
  for (int c = 0; c < 4; ++c)
#pragma unroll
    for (int i = 0; i < 8; ++i)
      Ltile[half * 8 + i][wave * 64 + c * 16 + col] = acc[c][i] * (1.0f / (float)TT);
  __syncthreads();

  const int row = tid >> 4, sub = tid & 15;
  float mx = -1e30f;
  for (int t = sub; t < TT; t += 16) mx = fmaxf(mx, Ltile[row][t]);
  red[row][sub] = mx;
  __syncthreads();
  if (sub == 0) {
    float m = red[row][0];
    for (int j = 1; j < 16; ++j) m = fmaxf(m, red[row][j]);
    red[row][0] = m;
  }
  __syncthreads();
  mx = red[row][0];
  __syncthreads();

  float sum = 0.f;
  for (int t = sub; t < TT; t += 16) {
    const float e = __expf(Ltile[row][t] - mx);
    Ltile[row][t] = e;
    sum += e;
  }
  red[row][sub] = sum;
  __syncthreads();
  if (sub == 0) {
    float ssum = 0.f;
    for (int j = 0; j < 16; ++j) ssum += red[row][j];
    red[row][0] = ssum;
  }
  __syncthreads();
  const float inv = 1.0f / red[row][0];

  __bf16* out = A2 + (size_t)s * TT * NN + (size_t)(nblk << 4) + row;  // A2[s][t, n]
  for (int t = sub; t < TT; t += 16)
    out[(size_t)t * NN] = (__bf16)(Ltile[row][t] * inv);
}

// ---------------------------------------------------------------------------
// LayerNorm over last dim C (f32 input), optional ReLU, f32 and/or bf16 out.
// One 256-thread block per row.
// ---------------------------------------------------------------------------
__global__ void ln_kernel(const float* __restrict__ X, const float* __restrict__ g,
                          const float* __restrict__ b, float* __restrict__ outf,
                          __bf16* __restrict__ outb, int C, int relu) {
  const int row = blockIdx.x, tid = threadIdx.x;
  const float* x = X + (size_t)row * C;
  float s1 = 0.f, s2 = 0.f;
  for (int c = tid; c < C; c += 256) { const float v = x[c]; s1 += v; s2 += v * v; }
  __shared__ float r1[256], r2[256];
  r1[tid] = s1; r2[tid] = s2;
  __syncthreads();
  for (int off = 128; off > 0; off >>= 1) {
    if (tid < off) { r1[tid] += r1[tid + off]; r2[tid] += r2[tid + off]; }
    __syncthreads();
  }
  const float mean = r1[0] / (float)C;
  const float var  = r2[0] / (float)C - mean * mean;
  const float inv  = rsqrtf(var + 1e-5f);
  for (int c = tid; c < C; c += 256) {
    float v = (x[c] - mean) * inv * g[c] + b[c];
    if (relu) v = fmaxf(v, 0.f);
    if (outf) outf[(size_t)row * C + c] = v;
    if (outb) outb[(size_t)row * C + c] = (__bf16)v;
  }
}

// Weight pack: f32 [In,Out] row-major -> bf16 W^T [Out,In] row-major.
__global__ void pack_wT(const float* __restrict__ W, __bf16* __restrict__ WT, int In, int Out) {
  const int i = blockIdx.x * 256 + threadIdx.x;
  if (i >= In * Out) return;
  const int o  = i % Out;
  const int in = i / Out;
  WT[(size_t)o * In + in] = (__bf16)W[i];
}

// Mean over T: H2 [S*T, 256] -> pooled [S, 256].
__global__ void pool_kernel(const float* __restrict__ H2, float* __restrict__ pooled) {
  const int s = blockIdx.x, c = threadIdx.x;   // 256 threads
  float acc = 0.f;
  for (int t = 0; t < TT; ++t) acc += H2[((size_t)s * TT + t) * (DD / 2) + c];
  pooled[s * (DD / 2) + c] = acc * (1.0f / (float)TT);
}

// Tiny dense f32 GEMV-ish (rows = 16 only): Y = X @ W + bias, W [In,Out] row-major.
__global__ void dense_f32(const float* __restrict__ X, const float* __restrict__ W,
                          const float* __restrict__ bias, float* __restrict__ Y,
                          int In, int Out) {
  const int row = blockIdx.x;
  for (int o = threadIdx.x; o < Out; o += blockDim.x) {
    float acc = bias[o];
    for (int k = 0; k < In; ++k) acc += X[(size_t)row * In + k] * W[(size_t)k * Out + o];
    Y[(size_t)row * Out + o] = acc;
  }
}

// Broadcast kqv_c [S, D] over T into out2 [S*T, D].
__global__ void bcast_kernel(const float* __restrict__ kqvc, float* __restrict__ out2) {
  const size_t i = (size_t)blockIdx.x * 256 + threadIdx.x;
  const size_t total = (size_t)SS * TT * DD;
  if (i >= total) return;
  const int d  = (int)(i % DD);
  const int st = (int)(i / DD);
  const int s  = st / TT;
  out2[i] = kqvc[(size_t)s * DD + d];
}

// ---------------------------------------------------------------------------
// Host launcher.
// ---------------------------------------------------------------------------
static inline size_t align256(size_t x) { return (x + 255) & ~(size_t)255; }

extern "C" void kernel_launch(void* const* d_in, const int* in_sizes, int n_in,
                              void* d_out, int out_size, void* d_ws, size_t ws_size,
                              hipStream_t stream) {
  (void)in_sizes; (void)n_in; (void)out_size; (void)ws_size;

  const float* share = (const float*)d_in[0];
  const float* fet   = (const float*)d_in[1];
  const float* ln_g  = (const float*)d_in[2];
  const float* ln_b  = (const float*)d_in[3];
  const float* qln_g = (const float*)d_in[4];
  const float* qln_b = (const float*)d_in[5];
  const float* Wq = (const float*)d_in[6];
  const float* Wk = (const float*)d_in[7];
  const float* Wv = (const float*)d_in[8];
  const float* Wm = (const float*)d_in[9];
  const float* bm = (const float*)d_in[10];
  const float* W1 = (const float*)d_in[11];
  const float* b1 = (const float*)d_in[12];
  const float* ln1_g = (const float*)d_in[13];
  const float* ln1_b = (const float*)d_in[14];
  const float* W2 = (const float*)d_in[15];
  const float* b2 = (const float*)d_in[16];
  const float* ln2_g = (const float*)d_in[17];
  const float* ln2_b = (const float*)d_in[18];
  const float* W3 = (const float*)d_in[19];
  const float* b3 = (const float*)d_in[20];
  const float* ln3_g = (const float*)d_in[21];
  const float* ln3_b = (const float*)d_in[22];
  const float* W4 = (const float*)d_in[23];
  const float* b4 = (const float*)d_in[24];

  const size_t SZ = (size_t)SS * TT * DD;   // 4,194,304
  float* out0 = (float*)d_out;              // kqv (post-mlp)
  float* out1 = out0 + SZ;                  // FeatureExtractTokens passthrough
  float* out2 = out1 + SZ;                  // kqv_c broadcast

  // Workspace carve-out.
  char* ws = (char*)d_ws;
  size_t off = 0;
  auto alloc = [&](size_t bytes) -> void* { void* p = ws + off; off = align256(off + bytes); return p; };
  __bf16* WqT  = (__bf16*)alloc((size_t)DD * DD * 2);
  __bf16* WkT  = (__bf16*)alloc((size_t)DD * DD * 2);
  __bf16* WvT  = (__bf16*)alloc((size_t)DD * DD * 2);
  __bf16* WmT  = (__bf16*)alloc((size_t)DD * DD * 2);
  __bf16* W1T  = (__bf16*)alloc((size_t)DD * DD * 2);
  __bf16* W2T  = (__bf16*)alloc((size_t)(DD / 2) * DD * 2);
  __bf16* sfb  = (__bf16*)alloc((size_t)NN * DD * 2);        // LN(share) bf16
  __bf16* Qin  = (__bf16*)alloc((size_t)SS * TT * DD * 2);   // LN(FET) bf16
  __bf16* Kb   = (__bf16*)alloc((size_t)NN * DD * 2);        // K [N,D]
  __bf16* Vt   = (__bf16*)alloc((size_t)DD * NN * 2);        // V^T [D,N]
  __bf16* Qb   = (__bf16*)alloc((size_t)SS * TT * DD * 2);   // Q [S*T,D]
  __bf16* A2   = (__bf16*)alloc((size_t)SS * TT * NN * 2);   // attn^T [S][T,N]
  __bf16* Kqvb = (__bf16*)alloc((size_t)SS * TT * DD * 2);   // kqv/T bf16
  __bf16* Mb   = (__bf16*)alloc((size_t)SS * TT * DD * 2);   // (kqv@Wm+bm) bf16
  float*  X1   = (float*) alloc((size_t)SS * TT * DD * 4);
  __bf16* H1b  = (__bf16*)alloc((size_t)SS * TT * DD * 2);
  float*  X2   = (float*) alloc((size_t)SS * TT * (DD / 2) * 4);
  float*  H2   = (float*) alloc((size_t)SS * TT * (DD / 2) * 4);
  float*  pooled = (float*)alloc((size_t)SS * (DD / 2) * 4);
  float*  X3   = (float*) alloc((size_t)SS * DD * 4);
  float*  gbuf = (float*) alloc((size_t)SS * DD * 4);
  float*  kqvc = (float*) alloc((size_t)SS * DD * 4);

  // 1) Pack weights to bf16 transposed.
  pack_wT<<<(DD * DD + 255) / 256, 256, 0, stream>>>(Wq, WqT, DD, DD);
  pack_wT<<<(DD * DD + 255) / 256, 256, 0, stream>>>(Wk, WkT, DD, DD);
  pack_wT<<<(DD * DD + 255) / 256, 256, 0, stream>>>(Wv, WvT, DD, DD);
  pack_wT<<<(DD * DD + 255) / 256, 256, 0, stream>>>(Wm, WmT, DD, DD);
  pack_wT<<<(DD * DD + 255) / 256, 256, 0, stream>>>(W1, W1T, DD, DD);
  pack_wT<<<(DD * (DD / 2) + 255) / 256, 256, 0, stream>>>(W2, W2T, DD, DD / 2);

  // 2) LayerNorms -> bf16 activations.
  ln_kernel<<<NN, 256, 0, stream>>>(share, ln_g, ln_b, nullptr, sfb, DD, 0);
  ln_kernel<<<SS * TT, 256, 0, stream>>>(fet, qln_g, qln_b, nullptr, Qin, DD, 0);

  // 3) Projections (WMMA NT GEMMs). Wave tile = 16x64.
  gemm_nt_bf16<<<dim3((NN / 16) * (DD / 64) / 8, 1), 256, 0, stream>>>(
      sfb, WkT, Kb, NN, DD, DD, 1.0f, 0, 0);
  gemm_nt_bf16_tc<<<(NN / 16) * (DD / 64) / 8, 256, 0, stream>>>(
      sfb, WvT, Vt, NN, DD, DD);
  gemm_nt_bf16<<<dim3((SS * TT / 16) * (DD / 64) / 8, 1), 256, 0, stream>>>(
      Qin, WqT, Qb, SS * TT, DD, DD, 1.0f, 0, 0);

  // 4) Logits + softmax over full T rows -> attn^T bf16.
  attn_softmax_kernel<<<dim3(NN / 16, SS), 256, 0, stream>>>(Kb, Qb, A2);

  // 5) kqv[s] = attn^T @ V / T  (batched over s, K = N = 8192).
  gemm_nt_bf16<<<dim3((TT / 16) * (DD / 64) / 8, SS), 256, 0, stream>>>(
      A2, Vt, Kqvb, TT, DD, NN, 1.0f / (float)TT,
      (size_t)TT * NN, (size_t)TT * DD);

  // 6) self.mlp: out0 = kqv @ Wm + bm (f32) + bf16 copy for the next GEMM.
  gemm_nt_f32bf16_bias<<<(SS * TT / 16) * (DD / 64) / 8, 256, 0, stream>>>(
      Kqvb, WmT, bm, out0, Mb, SS * TT, DD, DD);

  // 7) genQureyToken encode.
  gemm_nt_f32_bias<<<(SS * TT / 16) * (DD / 64) / 8, 256, 0, stream>>>(
      Mb, W1T, b1, X1, SS * TT, DD, DD);
  ln_kernel<<<SS * TT, 256, 0, stream>>>(X1, ln1_g, ln1_b, nullptr, H1b, DD, 1);
  gemm_nt_f32_bias<<<(SS * TT / 16) * ((DD / 2) / 64) / 8, 256, 0, stream>>>(
      H1b, W2T, b2, X2, SS * TT, DD / 2, DD);
  ln_kernel<<<SS * TT, 256, 0, stream>>>(X2, ln2_g, ln2_b, H2, nullptr, DD / 2, 1);

  // 8) Pool over T + decode tail (tiny: 16 rows).
  pool_kernel<<<SS, 256, 0, stream>>>(H2, pooled);
  dense_f32<<<SS, 256, 0, stream>>>(pooled, W3, b3, X3, DD / 2, DD);
  ln_kernel<<<SS, 256, 0, stream>>>(X3, ln3_g, ln3_b, gbuf, nullptr, DD, 1);
  dense_f32<<<SS, 256, 0, stream>>>(gbuf, W4, b4, kqvc, DD, DD);

  // 9) Outputs 1 & 2.
  hipMemcpyAsync(out1, fet, SZ * sizeof(float), hipMemcpyDeviceToDevice, stream);
  bcast_kernel<<<(int)((SZ + 255) / 256), 256, 0, stream>>>(kqvc, out2);
}
